// CustomRNN_81200651698796
// MI455X (gfx1250) — matrix operand order repaired and checked by reading, not compiled, not run on hardware
//
#include <hip/hip_runtime.h>

typedef __attribute__((ext_vector_type(16))) __bf16 bf16x16;
typedef __attribute__((ext_vector_type(8)))  float  f32x8;
typedef __attribute__((ext_vector_type(4)))  float  f32x4;

#define B_   512
#define S_   1024
#define I_   64
#define H_   256
#define C_   10
#define HROW 280   // 256 + 24 pad: 560B row stride == 12 dwords mod 64 -> bank-conflict-free b128

#if __has_builtin(__builtin_amdgcn_sched_barrier)
#define SCHED_FENCE() __builtin_amdgcn_sched_barrier(0)
#else
#define SCHED_FENCE()
#endif

__launch_bounds__(512, 1)
__global__ void rnn_fused_wmma(const float* __restrict__ inp,
                               const float* __restrict__ W_ih,
                               const float* __restrict__ b_ih,
                               const float* __restrict__ W_ho,
                               const float* __restrict__ b_ho,
                               float* __restrict__ out)
{
    // ping-pong h state: [buf][batch-local m (16)][hidden k (padded 280)] bf16, 2 x ~9 KB LDS
    __shared__ __bf16 sH[2][16 * HROW];

    const int tid  = threadIdx.x;
    const int lane = tid & 31;
    const int nt   = tid >> 5;      // wave id == 16-wide hidden output tile
    const int m    = lane & 15;     // A-row (hidden_out local) AND B/D column (batch local)
    const int kh   = lane >> 4;     // which K-half this lane holds
    const int b0   = blockIdx.x * 16;

    // ---- Preload A = W^T fragments into registers (reused 1024x) ----
    // A fragment (16x32 bf16), lane l elem e: row M = l&15, K = kk*32 + kh*16 + e
    const int colA = nt * 16 + m;   // hidden_out index of this lane's A row
    bf16x16 aWh[8];
    bf16x16 aWx[2];
    #pragma unroll
    for (int kk = 0; kk < 8; ++kk) {
        bf16x16 f;
        #pragma unroll
        for (int e = 0; e < 16; ++e) {
            int k = kk * 32 + kh * 16 + e;   // hidden_in
            f[e] = (__bf16)W_ih[(size_t)(I_ + k) * H_ + colA];      // Wh^T[colA][k]
        }
        aWh[kk] = f;
    }
    #pragma unroll
    for (int kk = 0; kk < 2; ++kk) {
        bf16x16 f;
        #pragma unroll
        for (int e = 0; e < 16; ++e) {
            int k = kk * 32 + kh * 16 + e;   // input dim
            f[e] = (__bf16)W_ih[(size_t)k * H_ + colA];             // Wx^T[colA][k]
        }
        aWx[kk] = f;
    }

    // bias per accumulator reg r: D row (hidden_out) = nt*16 + kh*8 + r
    f32x8 bias;
    #pragma unroll
    for (int r = 0; r < 8; ++r) bias[r] = b_ih[nt * 16 + kh * 8 + r];

    // h0 = 0 (buffer 0 is the first read buffer)
    for (int i = tid; i < 16 * HROW; i += 512) sH[0][i] = (__bf16)0.0f;
    __syncthreads();

    // x stream for this lane's batch row; lane holds K-half kh of the B fragment
    const float* xrow = inp + (size_t)(b0 + m) * S_ * I_ + kh * 16;
    // B fragment of h^T: lane l -> col N=batch=m, K = kk*32 + kh*16 + (0..15)
    const int hBoff = m * HROW + kh * 16;
    // D store: lane holds batch col m, hidden rows nt*16 + kh*8 + (0..7) -> one b128
    const int hSoff = m * HROW + nt * 16 + kh * 8;

    auto step = [&](int t, const __bf16* __restrict__ rd, __bf16* __restrict__ wr) {
        const float* xp = xrow + t * I_;
        if (nt == 0) __builtin_prefetch(xp + 2 * I_, 0, 3);   // global_prefetch_b8, 2 steps ahead

        // ---- issue ALL loads first: 8 global b128 (x) + 16 ds b128 (h) ----
        f32x4 xf[8];
        #pragma unroll
        for (int q = 0; q < 8; ++q)
            xf[q] = *(const f32x4*)(xp + ((q >> 2) * 32) + ((q & 3) * 4));

        bf16x16 bh[8];
        #pragma unroll
        for (int kk = 0; kk < 8; ++kk)
            bh[kk] = *(const bf16x16*)(rd + hBoff + kk * 32);

        // convert x to bf16 B-fragments while LDS loads are in flight
        bf16x16 bx[2];
        #pragma unroll
        for (int kk = 0; kk < 2; ++kk) {
            bf16x16 b;
            #pragma unroll
            for (int q = 0; q < 4; ++q) {
                #pragma unroll
                for (int e = 0; e < 4; ++e)
                    b[q * 4 + e] = (__bf16)xf[kk * 4 + q][e];
            }
            bx[kk] = b;
        }

        SCHED_FENCE();   // loads/converts may not sink below; WMMAs may not hoist above

        // ---- back-to-back WMMA chain (accumulator D->C chaining, no hazards) ----
        f32x8 acc = bias;
        #pragma unroll
        for (int kk = 0; kk < 2; ++kk)
            acc = __builtin_amdgcn_wmma_f32_16x16x32_bf16(
                      false, aWx[kk], false, bx[kk], (short)0, acc, false, false);
        #pragma unroll
        for (int kk = 0; kk < 8; ++kk)
            acc = __builtin_amdgcn_wmma_f32_16x16x32_bf16(
                      false, aWh[kk], false, bh[kk], (short)0, acc, false, false);

        // tanh (overflow-safe fast form) + write next h into the OTHER buffer
        #pragma unroll
        for (int r = 0; r < 8; ++r) {
            float v  = acc[r];
            float e2 = __expf(-2.0f * fabsf(v));
            float th = __fdividef(1.0f - e2, 1.0f + e2);
            wr[hSoff + r] = (__bf16)copysignf(th, v);
        }
        __syncthreads();   // single barrier per step: write-buffer complete for next step
    };

    for (int t = 0; t < S_; t += 2) {
        step(t,     sH[0], sH[1]);   // read buf0, write buf1
        step(t + 1, sH[1], sH[0]);   // read buf1, write buf0
    }
    // 1024 steps (even): final h lives in sH[0]

    // out = h_final @ W_ho + b_ho   (16 x 256 x 10 — negligible)
    if (tid < 16 * C_) {
        int mm = tid / C_, c = tid % C_;
        float s = b_ho[c];
        for (int k = 0; k < H_; ++k)
            s = fmaf((float)sH[0][mm * HROW + k], W_ho[k * C_ + c], s);
        out[(size_t)(b0 + mm) * C_ + c] = s;
    }
}

extern "C" void kernel_launch(void* const* d_in, const int* in_sizes, int n_in,
                              void* d_out, int out_size, void* d_ws, size_t ws_size,
                              hipStream_t stream) {
    (void)in_sizes; (void)n_in; (void)out_size; (void)d_ws; (void)ws_size;
    const float* inputs = (const float*)d_in[0];
    const float* W_ih   = (const float*)d_in[1];
    const float* b_ih   = (const float*)d_in[2];
    const float* W_ho   = (const float*)d_in[3];
    const float* b_ho   = (const float*)d_in[4];
    rnn_fused_wmma<<<dim3(B_ / 16), dim3(512), 0, stream>>>(
        inputs, W_ih, b_ih, W_ho, b_ho, (float*)d_out);
}
